// GCN_adj_31353261261177
// MI455X (gfx1250) — matrix operand order repaired and checked by reading, not compiled, not run on hardware
//
#include <hip/hip_runtime.h>
#include <hip/hip_bf16.h>
#include <stdint.h>

// ---------------------------------------------------------------------------
// Problem constants (from reference): B=256, N=400, CIN=H=128, COUT=16
// ---------------------------------------------------------------------------
#define B_    256
#define N_    400
#define H_    128
#define COUT_ 16
#define NCH   13            // ceil(400/32) 32-source chunks (mask word granularity)
#define NCH64 7             // ceil(400/64) 64-source chunks (mask kernel grid)
#define KPAD  416           // N padded to multiple of 32 for WMMA K loop
#define THRLOGIT 0.4054651081f   // ln(0.6/0.4): sigmoid(z)>0.6 <=> z>ln 1.5
#define EPS_  1e-5f

typedef __attribute__((ext_vector_type(16))) _Float16 v16h;
typedef __attribute__((ext_vector_type(8)))  _Float16 v8h;
typedef __attribute__((ext_vector_type(4)))  _Float16 v4h;
typedef __attribute__((ext_vector_type(8)))  float    v8f;

#define SHUF16(lo, hi) __builtin_shufflevector(lo, hi, 0,1,2,3,4,5,6,7,8,9,10,11,12,13,14,15)

static __device__ __forceinline__ float leaky02(float x) {
    return x > 0.f ? x : 0.2f * x;
}

// ---------------------------------------------------------------------------
// One-time: repack W_adj (f32 [400][400]) into f16 WMMA B-fragment layout:
// frag[dt][kk][lane][j], 16 contiguous halves per lane, zero-padded K>=400.
// ---------------------------------------------------------------------------
__global__ __launch_bounds__(32) void wadj_frag_kernel(
    const float* __restrict__ Wadj, _Float16* __restrict__ frag)
{
    const int kk = blockIdx.x, dt = blockIdx.y, lane = threadIdx.x;
    const int lh = lane >> 4, ll = lane & 15;
    const int d = dt * 16 + ll, kb = kk * 32;
    _Float16* p = frag + (((size_t)dt * NCH + kk) * 32 + lane) * 16;
    #pragma unroll
    for (int j = 0; j < 16; ++j) {
        int kg = kb + (j & 7) + ((j >> 3) << 4) + (lh << 3);
        p[j] = (kg < N_) ? (_Float16)Wadj[(size_t)kg * N_ + d] : (_Float16)0.f;
    }
}

// Same repack for a 128x128 weight matrix: frag[ct][kk][lane][j]
__global__ __launch_bounds__(32) void wmat_frag_kernel(
    const float* __restrict__ W, _Float16* __restrict__ frag)
{
    const int kk = blockIdx.x, ct = blockIdx.y, lane = threadIdx.x;
    const int lh = lane >> 4, ll = lane & 15;
    const int d = ct * 16 + ll, kb = kk * 32;
    _Float16* p = frag + (((size_t)ct * 4 + kk) * 32 + lane) * 16;
    #pragma unroll
    for (int j = 0; j < 16; ++j) {
        int kg = kb + (j & 7) + ((j >> 3) << 4) + (lh << 3);
        p[j] = (_Float16)W[(size_t)kg * 128 + d];
    }
}

// ---------------------------------------------------------------------------
// Kernel 1: mask[b,s,d] = (sigmoid(adj[b,s,:]@W_adj[:,d]+b_adj[d]) > 0.6)|(s==d)
// packed as bits over s into maskw[b][d][chunk32].
// Workgroup = (b, 64-source chunk): each B-fragment load now feeds 4 WMMAs
// (4 source tiles), doubling matrix work per load stall and halving total
// wadjfrag traffic. Produces two 32-bit mask words per d per workgroup.
// ---------------------------------------------------------------------------
__global__ __launch_bounds__(256) void mask_kernel(
    const float* __restrict__ adj, const _Float16* __restrict__ wfrag,
    const float* __restrict__ badj, uint32_t* __restrict__ maskw)
{
    __shared__ __align__(16) _Float16 As[64][KPAD];   // 53.2 KB
    const int b     = blockIdx.y;
    const int s0    = blockIdx.x * 64;
    const int tid   = threadIdx.x;
    const int wave  = __builtin_amdgcn_readfirstlane(tid >> 5);  // wave-uniform
    const int lane  = tid & 31;
    const int lh    = lane >> 4, ll = lane & 15;

    // Cooperative vectorized load + f32->f16 convert, zero-padded rows/cols
    for (int i = tid; i < 64 * 104; i += 256) {       // 104 float4 per row
        int r = i / 104, c4 = i - r * 104;
        int s = s0 + r;
        float4 v = make_float4(0.f, 0.f, 0.f, 0.f);
        if (s < N_ && c4 < 100)
            v = *(const float4*)(adj + ((size_t)b * N_ + s) * N_ + c4 * 4);
        union { uint2 u; _Float16 h[4]; } pk;
        pk.h[0] = (_Float16)v.x; pk.h[1] = (_Float16)v.y;
        pk.h[2] = (_Float16)v.z; pk.h[3] = (_Float16)v.w;
        *(uint2*)(&As[r][c4 * 4]) = pk.u;
    }
    __syncthreads();

    for (int dt = wave; dt < 25; dt += 8) {     // wave-uniform: EXEC stays full
        const int d0 = dt * 16;
        const int d  = d0 + ll;
        const _Float16* fb = wfrag + (((size_t)dt * NCH) * 32 + lane) * 16;
        // pull the next d-tile's fragment block toward this WGP
        if (dt + 8 < 25)
            __builtin_prefetch(wfrag + (((size_t)(dt + 8) * NCH) * 32 + lane) * 16, 0, 3);

        v8f cacc[4] = {};
        #pragma unroll
        for (int kk = 0; kk < NCH; ++kk) {
            const int kb = kk * 32;
            v16h bf = *(const v16h*)(fb + (size_t)kk * 512);
            #pragma unroll
            for (int t = 0; t < 4; ++t) {
                v8h alo = *(const v8h*)&As[t * 16 + ll][kb + (lh << 3)];
                v8h ahi = *(const v8h*)&As[t * 16 + ll][kb + 16 + (lh << 3)];
                v16h a = SHUF16(alo, ahi);
                cacc[t] = __builtin_amdgcn_wmma_f32_16x16x32_f16(false, a, false, bf, (short)0, cacc[t], false, false);
            }
        }
        // sigmoid(c + b_adj) > 0.6  <=>  c > ln(1.5) - b_adj
        const float thr = THRLOGIT - badj[d];
        unsigned by[4];
        #pragma unroll
        for (int t = 0; t < 4; ++t) {
            unsigned bb = 0;
            #pragma unroll
            for (int r = 0; r < 8; ++r)
                if (cacc[t][r] > thr) bb |= 1u << r;   // rows (s0 + 16t + r + 8*lh)
            by[t] = bb;
        }
        unsigned o0 = (unsigned)__shfl((int)by[0], ll + 16, 32);
        unsigned o1 = (unsigned)__shfl((int)by[1], ll + 16, 32);
        unsigned o2 = (unsigned)__shfl((int)by[2], ll + 16, 32);
        unsigned o3 = (unsigned)__shfl((int)by[3], ll + 16, 32);
        if (lh == 0) {
            const int cw0 = blockIdx.x * 2;            // 32-source chunk index
            unsigned w0 = by[0] | (o0 << 8) | (by[1] << 16) | (o1 << 24);
            unsigned w1 = by[2] | (o2 << 8) | (by[3] << 16) | (o3 << 24);
            if (d >= s0      && d < s0 + 32) w0 |= 1u << (d - s0);        // self loop
            if (d >= s0 + 32 && d < s0 + 64) w1 |= 1u << (d - s0 - 32);
            maskw[((size_t)b * N_ + d) * 16 + cw0] = w0;
            if (cw0 + 1 < NCH)
                maskw[((size_t)b * N_ + d) * 16 + cw0 + 1] = w1;
        }
    }
}

// ---------------------------------------------------------------------------
// Kernel 2: hp = in @ W   ([B*N,128] x [128,128] -> f16)
// 64-row chunk in LDS; W pre-packed as fragments (one v16h load each).
// ---------------------------------------------------------------------------
__global__ __launch_bounds__(256) void gemm_hp_kernel(
    const float* __restrict__ in, const _Float16* __restrict__ wfrag,
    _Float16* __restrict__ hp)
{
    __shared__ __align__(16) _Float16 Xs[64][136];
    const int tid  = threadIdx.x;
    const int wave = __builtin_amdgcn_readfirstlane(tid >> 5);
    const int lane = tid & 31;
    const int lh   = lane >> 4, ll = lane & 15;
    const size_t r0 = (size_t)blockIdx.x * 64;

    // Preload W fragments first so the loads overlap the LDS fill
    const _Float16* fb = wfrag + (((size_t)wave * 4) * 32 + lane) * 16;
    v16h bfs[4];
    #pragma unroll
    for (int kk = 0; kk < 4; ++kk)
        bfs[kk] = *(const v16h*)(fb + (size_t)kk * 512);

    for (int i = tid; i < 64 * 32; i += 256) {
        int r = i >> 5, c4 = i & 31;
        float4 v = *(const float4*)(in + (r0 + r) * 128 + c4 * 4);
        union { uint2 u; _Float16 h[4]; } pk;
        pk.h[0] = (_Float16)v.x; pk.h[1] = (_Float16)v.y;
        pk.h[2] = (_Float16)v.z; pk.h[3] = (_Float16)v.w;
        *(uint2*)(&Xs[r][c4 * 4]) = pk.u;
    }
    __syncthreads();

    const int h0 = wave * 16;
    v8f acc[4] = {};
    #pragma unroll
    for (int kk = 0; kk < 4; ++kk) {
        const int kb = kk * 32;
        #pragma unroll
        for (int rt = 0; rt < 4; ++rt) {
            v8h lo = *(const v8h*)&Xs[rt * 16 + ll][kb + (lh << 3)];
            v8h hi = *(const v8h*)&Xs[rt * 16 + ll][kb + 16 + (lh << 3)];
            v16h a = SHUF16(lo, hi);
            acc[rt] = __builtin_amdgcn_wmma_f32_16x16x32_f16(false, a, false, bfs[kk], (short)0, acc[rt], false, false);
        }
    }
    #pragma unroll
    for (int rt = 0; rt < 4; ++rt) {
        #pragma unroll
        for (int r = 0; r < 8; ++r) {
            size_t row = r0 + rt * 16 + r + (lh << 3);
            hp[row * 128 + h0 + ll] = (_Float16)acc[rt][r];
        }
    }
}

// ---------------------------------------------------------------------------
// Kernel 3: e_src = hp @ a_src, e_dst = hp @ a_dst (wave per node row)
// ---------------------------------------------------------------------------
__global__ __launch_bounds__(256) void edot_kernel(
    const _Float16* __restrict__ hp, const float* __restrict__ asrc,
    const float* __restrict__ adst, float* __restrict__ esrc,
    float* __restrict__ edst)
{
    const int tid  = threadIdx.x;
    const int wave = __builtin_amdgcn_readfirstlane(tid >> 5);
    const int lane = tid & 31;
    const size_t row = (size_t)blockIdx.x * 8 + wave;
    v4h v = *(const v4h*)(hp + row * 128 + lane * 4);
    float s1 = 0.f, s2 = 0.f;
    #pragma unroll
    for (int i = 0; i < 4; ++i) {
        int h = lane * 4 + i;
        float f = (float)v[i];
        s1 += f * asrc[h];
        s2 += f * adst[h];
    }
    #pragma unroll
    for (int off = 16; off > 0; off >>= 1) {
        s1 += __shfl_xor(s1, off, 32);
        s2 += __shfl_xor(s2, off, 32);
    }
    if (lane == 0) { esrc[row] = s1; edst[row] = s2; }
}

// ---------------------------------------------------------------------------
// Kernel 4: per (b, dst): online-softmax max & 1/sum over masked sources
// ---------------------------------------------------------------------------
__global__ __launch_bounds__(256) void stats_kernel(
    const float* __restrict__ esrc, const float* __restrict__ edst,
    const uint32_t* __restrict__ maskw,
    float* __restrict__ rowmax, float* __restrict__ rowinv)
{
    __shared__ float es[N_];
    const int b   = blockIdx.y;
    const int tid = threadIdx.x;
    for (int i = tid; i < N_; i += 256) es[i] = esrc[(size_t)b * N_ + i];
    __syncthreads();
    const int d = blockIdx.x * 256 + tid;
    if (d >= N_) return;
    const float ed = edst[(size_t)b * N_ + d];
    float m = -3.0e38f, sum = 0.f;
    for (int c = 0; c < NCH; ++c) {
        uint32_t w = maskw[((size_t)b * N_ + d) * 16 + c];
        const int nb = (c == NCH - 1) ? (N_ - 32 * (NCH - 1)) : 32;
        for (int i = 0; i < nb; ++i) {
            if ((w >> i) & 1u) {
                float sc = leaky02(es[c * 32 + i] + ed);
                if (sc > m) { sum = sum * __expf(m - sc) + 1.f; m = sc; }
                else        { sum += __expf(sc - m); }
            }
        }
    }
    rowmax[(size_t)b * N_ + d] = m;
    rowinv[(size_t)b * N_ + d] = 1.f / sum;   // self-loop => sum > 0
}

// ---------------------------------------------------------------------------
// Kernel 5: out[b,d,h] = sum_s alpha[s,d] * hp[b,s,h] + bias[h]
// Double-buffered LDS staging: fill(c+1) issues its global loads before the
// WMMA of chunk c, overlapping hp traffic with compute.
// ---------------------------------------------------------------------------
__global__ __launch_bounds__(256) void gat_agg_kernel(
    const _Float16* __restrict__ hp, const float* __restrict__ esrc,
    const float* __restrict__ edst, const float* __restrict__ rowmax,
    const float* __restrict__ rowinv, const uint32_t* __restrict__ maskw,
    const float* __restrict__ bias, float* __restrict__ out)
{
    __shared__ __align__(16) _Float16 HsT[2][128][40];   // [buf][h][s], padded
    __shared__ float esc[2][32];
    const int b    = blockIdx.y;
    const int d0   = blockIdx.x * 16;
    const int tid  = threadIdx.x;
    const int wave = __builtin_amdgcn_readfirstlane(tid >> 5);
    const int lane = tid & 31;
    const int lh   = lane >> 4, ll = lane & 15;
    const int h0   = wave * 16;
    const int d    = d0 + ll;

    const float ed  = edst[(size_t)b * N_ + d];
    const float mx  = rowmax[(size_t)b * N_ + d];
    const float inv = rowinv[(size_t)b * N_ + d];

    uint32_t wAll[NCH];
    #pragma unroll
    for (int c = 0; c < NCH; ++c)
        wAll[c] = maskw[((size_t)b * N_ + d) * 16 + c];

    auto fill = [&](int c) {
        const int bufi = c & 1;
        for (int i = tid; i < 32 * 64; i += 256) {
            int hcol = (i & 63) << 1, r = i >> 6;
            int s = c * 32 + r;
            uint32_t vv = 0u;
            if (s < N_) vv = *(const uint32_t*)(hp + ((size_t)b * N_ + s) * 128 + hcol);
            union { uint32_t u; _Float16 h[2]; } cv; cv.u = vv;
            HsT[bufi][hcol][r]     = cv.h[0];
            HsT[bufi][hcol + 1][r] = cv.h[1];
        }
        if (tid < 32) {
            int s = c * 32 + tid;
            esc[bufi][tid] = (s < N_) ? esrc[(size_t)b * N_ + s] : 0.f;
        }
    };

    fill(0);
    v8f acc = {};
    for (int c = 0; c < NCH; ++c) {
        __syncthreads();                 // buffer c ready for everyone
        if (c + 1 < NCH) fill(c + 1);    // overlap next chunk's loads with compute
        const int bufi = c & 1;
        const uint32_t w = wAll[c];
        v16h af;
        #pragma unroll
        for (int j = 0; j < 16; ++j) {
            int sk = (j & 7) + ((j >> 3) << 4) + (lh << 3);  // source within chunk
            int s  = c * 32 + sk;
            float al = 0.f;
            if (s < N_ && ((w >> sk) & 1u))
                al = __expf(leaky02(esc[bufi][sk] + ed) - mx) * inv;
            af[j] = (_Float16)al;
        }
        v8h blo = *(const v8h*)&HsT[bufi][h0 + ll][(lh << 3)];
        v8h bhi = *(const v8h*)&HsT[bufi][h0 + ll][16 + (lh << 3)];
        v16h bf = SHUF16(blo, bhi);
        acc = __builtin_amdgcn_wmma_f32_16x16x32_f16(false, af, false, bf, (short)0, acc, false, false);
    }
    #pragma unroll
    for (int r = 0; r < 8; ++r) {
        int dd = d0 + r + (lh << 3);
        out[((size_t)b * N_ + dd) * 128 + h0 + ll] = acc[r] + bias[h0 + ll];
    }
}

// ---------------------------------------------------------------------------
// Kernel 6: LayerNorm (+ optional ReLU), wave per node row, in place
// ---------------------------------------------------------------------------
__global__ __launch_bounds__(256) void ln_relu_kernel(
    float* __restrict__ x, const float* __restrict__ g,
    const float* __restrict__ bb, int do_relu)
{
    const int tid  = threadIdx.x;
    const int wave = __builtin_amdgcn_readfirstlane(tid >> 5);
    const int lane = tid & 31;
    const size_t row = (size_t)blockIdx.x * 8 + wave;
    float4 v = *(const float4*)(x + row * 128 + lane * 4);
    float s = v.x + v.y + v.z + v.w;
    #pragma unroll
    for (int off = 16; off > 0; off >>= 1) s += __shfl_xor(s, off, 32);
    const float mean = s * (1.f / 128.f);
    float q = (v.x - mean) * (v.x - mean) + (v.y - mean) * (v.y - mean)
            + (v.z - mean) * (v.z - mean) + (v.w - mean) * (v.w - mean);
    #pragma unroll
    for (int off = 16; off > 0; off >>= 1) q += __shfl_xor(q, off, 32);
    const float rinv = rsqrtf(q * (1.f / 128.f) + EPS_);
    const int h = lane * 4;
    float4 o;
    o.x = (v.x - mean) * rinv * g[h + 0] + bb[h + 0];
    o.y = (v.y - mean) * rinv * g[h + 1] + bb[h + 1];
    o.z = (v.z - mean) * rinv * g[h + 2] + bb[h + 2];
    o.w = (v.w - mean) * rinv * g[h + 3] + bb[h + 3];
    if (do_relu) {
        o.x = fmaxf(o.x, 0.f); o.y = fmaxf(o.y, 0.f);
        o.z = fmaxf(o.z, 0.f); o.w = fmaxf(o.w, 0.f);
    }
    *(float4*)(x + row * 128 + h) = o;
}

// ---------------------------------------------------------------------------
// Kernel 7: global mean pool over nodes + final linear -> [B, 16]
// ---------------------------------------------------------------------------
__global__ __launch_bounds__(128) void pool_lin_kernel(
    const float* __restrict__ h3, const float* __restrict__ Wlin,
    const float* __restrict__ blin, float* __restrict__ out)
{
    __shared__ float gbuf[128];
    const int b = blockIdx.x, tid = threadIdx.x;
    float acc = 0.f;
    for (int n = 0; n < N_; ++n)
        acc += h3[((size_t)b * N_ + n) * 128 + tid];
    gbuf[tid] = acc * (1.f / N_);
    __syncthreads();
    if (tid < COUT_) {
        float o = blin[tid];
        for (int h = 0; h < 128; ++h)
            o += gbuf[h] * Wlin[h * COUT_ + tid];
        out[b * COUT_ + tid] = o;
    }
}

// ---------------------------------------------------------------------------
extern "C" void kernel_launch(void* const* d_in, const int* in_sizes, int n_in,
                              void* d_out, int out_size, void* d_ws, size_t ws_size,
                              hipStream_t stream)
{
    const float* x     = (const float*)d_in[0];
    const float* adj   = (const float*)d_in[1];
    // d_in[2] = edge (unused), d_in[3] = batch (unused; dense mean over axis 1)
    const float* Wadj  = (const float*)d_in[4];
    const float* badj  = (const float*)d_in[5];
    const float* W1    = (const float*)d_in[6];
    const float* asrc1 = (const float*)d_in[7];
    const float* adst1 = (const float*)d_in[8];
    const float* b1    = (const float*)d_in[9];
    const float* W2    = (const float*)d_in[10];
    const float* asrc2 = (const float*)d_in[11];
    const float* adst2 = (const float*)d_in[12];
    const float* b2    = (const float*)d_in[13];
    const float* lng   = (const float*)d_in[14];
    const float* lnb   = (const float*)d_in[15];
    const float* Wlin  = (const float*)d_in[16];
    const float* blin  = (const float*)d_in[17];
    float* out = (float*)d_out;

    char* ws = (char*)d_ws;
    size_t off = 0;
    uint32_t* maskw     = (uint32_t*)(ws + off); off += (size_t)B_ * N_ * 16 * 4;      // 6.5 MB
    _Float16* wadjfrag  = (_Float16*)(ws + off); off += (size_t)25 * NCH * 512 * 2;    // 333 KB
    _Float16* wf1       = (_Float16*)(ws + off); off += (size_t)8 * 4 * 512 * 2;       // 32 KB
    _Float16* wf2       = (_Float16*)(ws + off); off += (size_t)8 * 4 * 512 * 2;       // 32 KB
    _Float16* hp        = (_Float16*)(ws + off); off += (size_t)B_ * N_ * H_ * 2;      // 26 MB
    float* esrc         = (float*)(ws + off);    off += (size_t)B_ * N_ * 4;
    float* edst         = (float*)(ws + off);    off += (size_t)B_ * N_ * 4;
    float* rmax         = (float*)(ws + off);    off += (size_t)B_ * N_ * 4;
    float* rinv         = (float*)(ws + off);    off += (size_t)B_ * N_ * 4;
    float* bufA         = (float*)(ws + off);    off += (size_t)B_ * N_ * H_ * 4;      // 52 MB
    float* bufB         = (float*)(ws + off);    off += (size_t)B_ * N_ * H_ * 4;      // 52 MB
    (void)ws_size; (void)in_sizes; (void)n_in; (void)out_size;

    const int rows = B_ * N_;   // 102400

    // One-time weight repacks into WMMA fragment layout
    wadj_frag_kernel<<<dim3(NCH, 25), 32, 0, stream>>>(Wadj, wadjfrag);
    wmat_frag_kernel<<<dim3(4, 8), 32, 0, stream>>>(W1, wf1);
    wmat_frag_kernel<<<dim3(4, 8), 32, 0, stream>>>(W2, wf2);

    // Edge mask (shared by all 3 GAT layers); 64-source chunks
    mask_kernel<<<dim3(NCH64, B_), 256, 0, stream>>>(adj, wadjfrag, badj, maskw);

    // ---- GAT layer 1 ----
    gemm_hp_kernel<<<rows / 64, 256, 0, stream>>>(x, wf1, hp);
    edot_kernel<<<rows / 8, 256, 0, stream>>>(hp, asrc1, adst1, esrc, edst);
    stats_kernel<<<dim3(2, B_), 256, 0, stream>>>(esrc, edst, maskw, rmax, rinv);
    gat_agg_kernel<<<dim3(25, B_), 256, 0, stream>>>(hp, esrc, edst, rmax, rinv, maskw, b1, bufA);
    ln_relu_kernel<<<rows / 8, 256, 0, stream>>>(bufA, lng, lnb, 1);

    // ---- GAT layer 2 ----
    gemm_hp_kernel<<<rows / 64, 256, 0, stream>>>(bufA, wf2, hp);
    edot_kernel<<<rows / 8, 256, 0, stream>>>(hp, asrc2, adst2, esrc, edst);
    stats_kernel<<<dim3(2, B_), 256, 0, stream>>>(esrc, edst, maskw, rmax, rinv);
    gat_agg_kernel<<<dim3(25, B_), 256, 0, stream>>>(hp, esrc, edst, rmax, rinv, maskw, b2, bufB);
    ln_relu_kernel<<<rows / 8, 256, 0, stream>>>(bufB, lng, lnb, 1);

    // ---- GAT layer 3 (conv2 applied twice) ----
    gemm_hp_kernel<<<rows / 64, 256, 0, stream>>>(bufB, wf2, hp);
    edot_kernel<<<rows / 8, 256, 0, stream>>>(hp, asrc2, adst2, esrc, edst);
    stats_kernel<<<dim3(2, B_), 256, 0, stream>>>(esrc, edst, maskw, rmax, rinv);
    gat_agg_kernel<<<dim3(25, B_), 256, 0, stream>>>(hp, esrc, edst, rmax, rinv, maskw, b2, bufA);

    // ---- mean pool + linear head ----
    pool_lin_kernel<<<B_, 128, 0, stream>>>(bufA, Wlin, blin, out);
}
